// GeneAwareMLP_36764920054438
// MI455X (gfx1250) — matrix operand order repaired and checked by reading, not compiled, not run on hardware
//
#include <hip/hip_runtime.h>
#include <hip/hip_bf16.h>
#include <math.h>

// Problem constants (fixed by the reference)
#define BSZ   256
#define D_IN  5000
#define KP1   5024          // D_IN padded up to a multiple of 32 for the K-loop
#define H1    2048
#define H2    1024
#define ISO   100000
#define NGENE 25000
#define EPS_LN 1e-5f
#define EPS_SM 1e-8f

typedef __attribute__((ext_vector_type(8)))  float  v8f;
typedef __attribute__((ext_vector_type(8)))  __bf16 v8bf;
typedef __attribute__((ext_vector_type(16))) __bf16 v16bf;
typedef __attribute__((ext_vector_type(4)))  unsigned int u32x4;
typedef __attribute__((ext_vector_type(8)))  int i32x8;
typedef __attribute__((ext_vector_type(4)))  int i32x4;

// ---------------------------------------------------------------------------
// Convert x (fp32, [256 x 5000]) to bf16 padded to [256 x 5024] (zeros in tail)
// ---------------------------------------------------------------------------
__global__ __launch_bounds__(256)
void cvt_pad_x_kernel(const float* __restrict__ x, __bf16* __restrict__ xb) {
  const int total = BSZ * KP1;
  for (int i = blockIdx.x * blockDim.x + threadIdx.x; i < total;
       i += gridDim.x * blockDim.x) {
    int r = i / KP1;
    int c = i - r * KP1;
    float v = (c < D_IN) ? x[(size_t)r * D_IN + c] : 0.0f;
    xb[i] = (__bf16)v;
  }
}

// ---------------------------------------------------------------------------
// TDM: DMA one B tile (32 rows x BN cols, fp32) from global into LDS.
// D# per CDNA5 ISA:
//  - group0: count=1, lds_addr, global_addr (bytes), type=2
//  - group1: data_size=4B, pad_enable (row stride BN+2 dwords in LDS),
//            tensor_dim0=N, tensor_dim1=K-kb (OOB rows read as 0 -> K tail),
//            tile_dim0=BN, tile_dim1=32, dim0_stride=N
//  - group2/3: dims 2..4 = 1 (single 2D slice), strides 0
//  - 5th data arg (clang-23 6-arg form): VADDR4 is unused per ISA -> zeros
// Tracked on TENSORcnt -> independent of the WMMA loadcnt/dscnt waits.
// ---------------------------------------------------------------------------
template <int BN>
__device__ __forceinline__ void tdm_load_b_tile(const float* B, int K, int N,
                                                int kb, int nbase,
                                                unsigned lds_off) {
  const unsigned long long gaddr =
      (unsigned long long)(uintptr_t)(B + (size_t)kb * N + nbase);
  const int td1 = K - kb;                        // remaining valid rows
  constexpr unsigned pi = (BN == 32) ? 4u : 3u;  // pad interval = BN dwords
  u32x4 g0;
  g0[0] = 1u;                                               // count=1
  g0[1] = lds_off;                                          // LDS byte addr
  g0[2] = (unsigned)gaddr;                                  // global[31:0]
  g0[3] = (unsigned)((gaddr >> 32) & 0x01FFFFFFu) | (2u << 30);  // type=2
  i32x8 g1;
  g1[0] = (int)((2u << 16) | (1u << 20) | (pi << 22) | (1u << 25));
  g1[1] = (int)(((unsigned)N & 0xFFFFu) << 16);             // dim0 lo16
  g1[2] = (int)((((unsigned)N >> 16) & 0xFFFFu) |           // dim0 hi16
                (((unsigned)td1 & 0xFFFFu) << 16));         // dim1 lo16
  g1[3] = (int)((((unsigned)td1 >> 16) & 0xFFFFu) |         // dim1 hi16
                ((unsigned)BN << 16));                      // tile_dim0
  g1[4] = 32;                                               // tile_dim1=32
  g1[5] = N;                                                // dim0_stride lo32
  g1[6] = 0;
  g1[7] = 0;
  i32x4 g2; g2[0] = 1; g2[1] = 1; g2[2] = 0; g2[3] = 0;     // dims2/3 = 1
  i32x4 g3; g3[0] = 0; g3[1] = 0x10000; g3[2] = 0; g3[3] = 0; // dim4 = 1
  i32x8 g4 = {0, 0, 0, 0, 0, 0, 0, 0};                      // VADDR4 unused
  __builtin_amdgcn_tensor_load_to_lds(g0, g1, g2, g3, g4, 0);
}

// ---------------------------------------------------------------------------
// WMMA GEMM: C[256, N] = A[256 x K](bf16, stride lda) * B[K x N](fp32) + bias
// Block = 256 threads = 8 waves; block covers all 256 rows x (NT*16) cols.
// Wave w owns M-tiles 2w,2w+1. B tiles (32 x BN fp32) are DMA'd by the TDM
// into double-buffered LDS (padded row stride BN+2 -> bank-conflict-free
// fragment gathers); fp32->bf16 conversion happens at fragment build.
// Requires N % BN == 0 (2048 / 1024 / 100000 all satisfy this).
// ---------------------------------------------------------------------------
template <int NT>
__global__ __launch_bounds__(256)
void wmma_gemm_bf16_kernel(const __bf16* __restrict__ A, int lda, int Ksteps,
                           const float* __restrict__ B, int K, int N,
                           const float* __restrict__ bias,
                           float* __restrict__ C) {
  constexpr int BN  = NT * 16;       // columns per block
  constexpr int BNP = BN + 2;        // padded LDS row stride (dwords)
  __shared__ __align__(16) float BsF[2][32 * BNP];

  const int nbase = blockIdx.x * BN;
  const int t     = threadIdx.x;
  const int wave  = t >> 5;          // 0..7 (wave32)
  const int lane  = t & 31;
  const int lhalf = lane >> 4;       // 0 | 1
  const int l16   = lane & 15;

  const int m0 = wave * 32;          // two M-tiles per wave
  const int m1 = m0 + 16;

  v8f acc0[NT] = {};
  v8f acc1[NT] = {};

  // ISA A-layout (16-bit A 16x32): lanes 0-15 hold K 0..7 & 16..23,
  // lanes 16-31 hold K 8..15 & 24..31, M = lane%16.
  const int    koff  = lhalf * 8;
  const size_t arow0 = (size_t)(m0 + l16) * lda;
  const size_t arow1 = (size_t)(m1 + l16) * lda;

  const unsigned lds0 = (unsigned)(uintptr_t)&BsF[0][0];
  const unsigned lds1 = (unsigned)(uintptr_t)&BsF[1][0];

  if (wave == 0) {                   // one wave per block drives the TDM
    tdm_load_b_tile<BN>(B, K, N, 0, nbase, lds0);
    __builtin_amdgcn_s_wait_tensorcnt(0);
  }
  __syncthreads();

  for (int ks = 0; ks < Ksteps; ++ks) {
    const int kb  = ks * 32;
    const int buf = ks & 1;
    if (wave == 0 && (ks + 1) < Ksteps)       // prefetch next tile (TENSORcnt)
      tdm_load_b_tile<BN>(B, K, N, kb + 32, nbase, buf ? lds0 : lds1);

    // A fragments: two 16B loads per M-tile, reused across all N-tiles.
    const __bf16* ap0 = A + arow0 + kb + koff;
    const __bf16* ap1 = A + arow1 + kb + koff;
    v8bf a0lo = *(const v8bf*)(ap0);
    v8bf a0hi = *(const v8bf*)(ap0 + 16);
    v8bf a1lo = *(const v8bf*)(ap1);
    v8bf a1hi = *(const v8bf*)(ap1 + 16);
    v16bf af0, af1;
    #pragma unroll
    for (int e = 0; e < 8; ++e) {
      af0[e] = a0lo[e]; af0[e + 8] = a0hi[e];
      af1[e] = a1lo[e]; af1[e + 8] = a1hi[e];
    }

    // B fragments from LDS [k][n] fp32 (row stride BNP): lane -> column
    // n = nt*16+l16, K = lhalf*8 + (0..7) and 16 + lhalf*8 + (0..7).
    const float* bbase = &BsF[buf][lhalf * 8 * BNP];
    #pragma unroll
    for (int nt = 0; nt < NT; ++nt) {
      const float* bp = bbase + nt * 16 + l16;
      v16bf bfrag;
      #pragma unroll
      for (int e = 0; e < 16; ++e) bfrag[e] = (__bf16)bp[(e + (e & 8)) * BNP];

      acc0[nt] = __builtin_amdgcn_wmma_f32_16x16x32_bf16(
          false, af0, false, bfrag, (short)0, acc0[nt], false, false);
      acc1[nt] = __builtin_amdgcn_wmma_f32_16x16x32_bf16(
          false, af1, false, bfrag, (short)0, acc1[nt], false, false);
    }

    if (wave == 0) __builtin_amdgcn_s_wait_tensorcnt(0);
    __syncthreads();
  }

  // Store: C/D layout -> VGPR v holds M = mtile + v + 8*lhalf, N = lane%16.
  #pragma unroll
  for (int nt = 0; nt < NT; ++nt) {
    const int gn = nbase + nt * 16 + l16;
    const float bv = bias ? bias[gn] : 0.0f;
    #pragma unroll
    for (int v = 0; v < 8; ++v) {
      int mm = m0 + v + lhalf * 8;
      C[(size_t)mm * N + gn] = acc0[nt][v] + bv;
      mm = m1 + v + lhalf * 8;
      C[(size_t)mm * N + gn] = acc1[nt][v] + bv;
    }
  }
}

// ---------------------------------------------------------------------------
// Per-row LayerNorm (biased var, eps=1e-5) + exact-erf GELU, fp32 -> bf16
// ---------------------------------------------------------------------------
__global__ __launch_bounds__(256)
void ln_gelu_kernel(const float* __restrict__ X, int H,
                    const float* __restrict__ g, const float* __restrict__ beta,
                    __bf16* __restrict__ Y) {
  __shared__ float red0[8];
  __shared__ float red1[8];
  const int row = blockIdx.x;
  const float* x = X + (size_t)row * H;

  float s = 0.0f, s2 = 0.0f;
  for (int i = threadIdx.x; i < H; i += 256) {
    float v = x[i];
    s += v; s2 += v * v;
  }
  for (int off = 16; off > 0; off >>= 1) {
    s  += __shfl_down(s,  off, 32);
    s2 += __shfl_down(s2, off, 32);
  }
  const int wave = threadIdx.x >> 5, lane = threadIdx.x & 31;
  if (lane == 0) { red0[wave] = s; red1[wave] = s2; }
  __syncthreads();
  if (threadIdx.x == 0) {
    float ts = 0.0f, ts2 = 0.0f;
    #pragma unroll
    for (int w = 0; w < 8; ++w) { ts += red0[w]; ts2 += red1[w]; }
    float mu  = ts / (float)H;
    float var = ts2 / (float)H - mu * mu;
    red0[0] = mu;
    red1[0] = rsqrtf(var + EPS_LN);
  }
  __syncthreads();
  const float mu = red0[0], rstd = red1[0];
  __bf16* y = Y + (size_t)row * H;
  for (int i = threadIdx.x; i < H; i += 256) {
    float v  = (x[i] - mu) * rstd * g[i] + beta[i];
    float ge = 0.5f * v * (1.0f + erff(v * 0.70710678118654752f));
    y[i] = (__bf16)ge;
  }
}

// ---------------------------------------------------------------------------
// Row max over logits [256 x ISO]; one block per row.
// ---------------------------------------------------------------------------
__global__ __launch_bounds__(256)
void row_max_kernel(const float* __restrict__ L, float* __restrict__ rmax) {
  __shared__ float red[8];
  const int row = blockIdx.x;
  const float* x = L + (size_t)row * ISO;
  float m = -INFINITY;
  for (int i = threadIdx.x; i < ISO; i += 256) m = fmaxf(m, x[i]);
  for (int off = 16; off > 0; off >>= 1) m = fmaxf(m, __shfl_down(m, off, 32));
  const int wave = threadIdx.x >> 5, lane = threadIdx.x & 31;
  if (lane == 0) red[wave] = m;
  __syncthreads();
  if (threadIdx.x == 0) {
    float tmax = -INFINITY;
    #pragma unroll
    for (int w = 0; w < 8; ++w) tmax = fmaxf(tmax, red[w]);
    rmax[row] = tmax;
  }
}

__global__ __launch_bounds__(256)
void zero_f32_kernel(float* __restrict__ p, int n) {
  for (int i = blockIdx.x * blockDim.x + threadIdx.x; i < n;
       i += gridDim.x * blockDim.x)
    p[i] = 0.0f;
}

// exp(logit - rowmax) in place + per-gene segment sum via fp32 global atomics
__global__ __launch_bounds__(256)
void exp_segsum_kernel(float* __restrict__ logits, const float* __restrict__ rmax,
                       const int* __restrict__ idx, float* __restrict__ denom) {
  const int total = BSZ * ISO;
  for (int i = blockIdx.x * blockDim.x + threadIdx.x; i < total;
       i += gridDim.x * blockDim.x) {
    int row = i / ISO;
    int col = i - row * ISO;
    float e = expf(logits[i] - rmax[row]);
    logits[i] = e;
    unsafeAtomicAdd(&denom[(size_t)row * NGENE + idx[col]], e);
  }
}

__global__ __launch_bounds__(256)
void normalize_kernel(float* __restrict__ out, const float* __restrict__ denom,
                      const int* __restrict__ idx) {
  const int total = BSZ * ISO;
  for (int i = blockIdx.x * blockDim.x + threadIdx.x; i < total;
       i += gridDim.x * blockDim.x) {
    int row = i / ISO;
    int col = i - row * ISO;
    float d = fmaxf(denom[(size_t)row * NGENE + idx[col]], EPS_SM);
    out[i] = out[i] / d;
  }
}

// ---------------------------------------------------------------------------
extern "C" void kernel_launch(void* const* d_in, const int* in_sizes, int n_in,
                              void* d_out, int out_size, void* d_ws, size_t ws_size,
                              hipStream_t stream) {
  (void)in_sizes; (void)n_in; (void)out_size; (void)ws_size;

  const float* x     = (const float*)d_in[0];
  const int*   idx   = (const int*)  d_in[1];
  const float* W1    = (const float*)d_in[2];
  const float* b1    = (const float*)d_in[3];
  const float* g1    = (const float*)d_in[4];
  const float* beta1 = (const float*)d_in[5];
  const float* W2    = (const float*)d_in[6];
  const float* b2    = (const float*)d_in[7];
  const float* g2    = (const float*)d_in[8];
  const float* beta2 = (const float*)d_in[9];
  const float* W3    = (const float*)d_in[10];
  const float* b3    = (const float*)d_in[11];
  float* out = (float*)d_out;

  // Workspace carve-up (all offsets 256B-aligned); total ~31.4 MB.
  char* ws = (char*)d_ws;
  __bf16* xb     = (__bf16*)(ws);                                   // 256*5024*2
  float*  h1raw  = (float*) (ws + 2572288);                         // 256*2048*4
  __bf16* h1b    = (__bf16*)(ws + 2572288 + 2097152);               // 256*2048*2
  float*  h2raw  = (float*) (ws + 2572288 + 2097152 + 1048576);     // 256*1024*4
  __bf16* h2b    = (__bf16*)(ws + 2572288 + 2097152 + 1048576 + 1048576);
  float*  rmax   = (float*) (ws + 2572288 + 2097152 + 1048576 + 1048576 + 524288);
  float*  denom  = (float*) (ws + 2572288 + 2097152 + 1048576 + 1048576 + 524288 + 1024);

  // 1) x -> bf16, padded K
  cvt_pad_x_kernel<<<2048, 256, 0, stream>>>(x, xb);

  // 2) h1raw = x @ W1 + b1   (K=5000 -> 157 K-steps, A stride 5024)
  wmma_gemm_bf16_kernel<1><<<H1 / 16, 256, 0, stream>>>(xb, KP1, KP1 / 32,
                                                        W1, D_IN, H1, b1, h1raw);
  // 3) LN + GELU -> bf16
  ln_gelu_kernel<<<BSZ, 256, 0, stream>>>(h1raw, H1, g1, beta1, h1b);

  // 4) h2raw = h1 @ W2 + b2
  wmma_gemm_bf16_kernel<1><<<H2 / 16, 256, 0, stream>>>(h1b, H1, H1 / 32,
                                                        W2, H1, H2, b2, h2raw);
  // 5) LN + GELU -> bf16
  ln_gelu_kernel<<<BSZ, 256, 0, stream>>>(h2raw, H2, g2, beta2, h2b);

  // 6) logits = h2 @ W3 + b3 -> d_out  (32 cols/block, 3125 blocks)
  wmma_gemm_bf16_kernel<2><<<ISO / 32, 256, 0, stream>>>(h2b, H2, H2 / 32,
                                                         W3, H2, ISO, b3, out);

  // 7) per-row max
  row_max_kernel<<<BSZ, 256, 0, stream>>>(out, rmax);

  // 8) zero per-gene denominators, then exp + segment-sum (atomics)
  zero_f32_kernel<<<4096, 256, 0, stream>>>(denom, BSZ * NGENE);
  exp_segsum_kernel<<<8192, 256, 0, stream>>>(out, rmax, idx, denom);

  // 9) gather denom per isoform and normalize
  normalize_kernel<<<8192, 256, 0, stream>>>(out, denom, idx);
}
